// CausalSelfAttention_25443386261518
// MI455X (gfx1250) — matrix operand order repaired and checked by reading, not compiled, not run on hardware
//
#include <hip/hip_runtime.h>
#include <hip/hip_bf16.h>
#include <math.h>

typedef __attribute__((ext_vector_type(16))) _Float16 v16h;
typedef __attribute__((ext_vector_type(8)))  _Float16 v8h;
typedef __attribute__((ext_vector_type(8)))  float    v8f;

#define TT 4096
#define CC 1024
#define NH 16
#define HD 64

// ---------------------------------------------------------------------------
// WMMA fragment loaders for v_wmma_f32_16x16x32_f16 (CDNA5 ISA 7.12.2 layouts)
// ---------------------------------------------------------------------------
static __device__ inline v16h frag_from2(const _Float16* p0, const _Float16* p1) {
  v8h lo = *(const v8h*)p0;   // 16 bytes
  v8h hi = *(const v8h*)p1;   // 16 bytes
  v16h r;
#pragma unroll
  for (int i = 0; i < 8; ++i) { r[i] = lo[i]; r[i + 8] = hi[i]; }
  return r;
}

// A fragment 16x32 (MxK), A row-major [M][K]:
//   lane 0-15:  M=lane,    elems 0-7 -> K=k0+0..7,   elems 8-15 -> K=k0+16..23
//   lane 16-31: M=lane-16, elems 0-7 -> K=k0+8..15,  elems 8-15 -> K=k0+24..31
static __device__ inline v16h load_a_frag(const _Float16* A, int lda, int m0, int k0, int lane) {
  const _Float16* p = A + (size_t)(m0 + (lane & 15)) * lda + k0 + ((lane >> 4) << 3);
  return frag_from2(p, p + 16);
}

// B fragment 32x16 (KxN) read from B^T stored [N][K] (row n contiguous in K):
//   lane 0-15:  N=lane,    elems 0-7 -> K=k0+0..7,   elems 8-15 -> K=k0+16..23
//   lane 16-31: N=lane-16, elems 0-7 -> K=k0+8..15,  elems 8-15 -> K=k0+24..31
static __device__ inline v16h load_b_frag(const _Float16* Bt, int ldb, int n0, int k0, int lane) {
  const _Float16* p = Bt + (size_t)(n0 + (lane & 15)) * ldb + k0 + ((lane >> 4) << 3);
  return frag_from2(p, p + 16);
}

static __device__ inline v8f wmma_f16(v16h a, v16h b, v8f c) {
  return __builtin_amdgcn_wmma_f32_16x16x32_f16(false, a, false, b, (short)0, c, false, false);
}

// CDNA5 async global->LDS copy (per-lane 16B), tracked by ASYNCcnt.
static __device__ inline void async_b128(unsigned lds_off, const void* gaddr) {
  asm volatile("global_load_async_to_lds_b128 %0, %1, off"
               :: "v"(lds_off), "v"(gaddr) : "memory");
}

// ---------------------------------------------------------------------------
// Generic GEMM: C[M][N] = A[M][K](f16) * B[K][N] + bias,  B given as Bt[N][K]
// Block: 256 threads = 8 waves; block tile 128(M) x 128(N); wave tile 32x64.
// Per K-step: 6 fragment loads -> 8 WMMAs (0.75 b128-pairs per WMMA).
// ---------------------------------------------------------------------------
__global__ __launch_bounds__(256)
void gemm_bias_f16(const _Float16* __restrict__ A, const _Float16* __restrict__ Bt,
                   const float* __restrict__ bias, float* __restrict__ C,
                   int M, int N, int K) {
  const int lane = threadIdx.x & 31;
  const int w    = threadIdx.x >> 5;
  const int m0   = blockIdx.x * 128 + (w & 3) * 32;   // 4 waves across M
  const int n0   = blockIdx.y * 128 + (w >> 2) * 64;  // 2 waves across N

  v8f c[2][4] = {};
  for (int k0 = 0; k0 < K; k0 += 32) {
    v16h a0 = load_a_frag(A, K, m0,      k0, lane);
    v16h a1 = load_a_frag(A, K, m0 + 16, k0, lane);
    v16h b0 = load_b_frag(Bt, K, n0,      k0, lane);
    v16h b1 = load_b_frag(Bt, K, n0 + 16, k0, lane);
    v16h b2 = load_b_frag(Bt, K, n0 + 32, k0, lane);
    v16h b3 = load_b_frag(Bt, K, n0 + 48, k0, lane);
    c[0][0] = wmma_f16(a0, b0, c[0][0]);
    c[0][1] = wmma_f16(a0, b1, c[0][1]);
    c[0][2] = wmma_f16(a0, b2, c[0][2]);
    c[0][3] = wmma_f16(a0, b3, c[0][3]);
    c[1][0] = wmma_f16(a1, b0, c[1][0]);
    c[1][1] = wmma_f16(a1, b1, c[1][1]);
    c[1][2] = wmma_f16(a1, b2, c[1][2]);
    c[1][3] = wmma_f16(a1, b3, c[1][3]);
  }

  const int nl = lane & 15;
  const int mh = (lane >> 4) << 3;   // C-layout: VGPR r -> row r + 8*(lane/16)
#pragma unroll
  for (int i = 0; i < 2; ++i)
#pragma unroll
    for (int j = 0; j < 4; ++j)
#pragma unroll
      for (int r = 0; r < 8; ++r) {
        int row = m0 + 16 * i + mh + r;
        int col = n0 + 16 * j + nl;
        C[(size_t)row * N + col] = c[i][j][r] + bias[col];
      }
}

// ---------------------------------------------------------------------------
// Convert / transpose-convert helpers (bandwidth-trivial)
// ---------------------------------------------------------------------------
__global__ __launch_bounds__(256)
void convert_f16(const float* __restrict__ X, _Float16* __restrict__ Xh, int n) {
  int idx = blockIdx.x * 256 + threadIdx.x;
  if (idx < n) Xh[idx] = (_Float16)X[idx];
}

__global__ __launch_bounds__(256)
void convert_tr_f16(const float* __restrict__ W, _Float16* __restrict__ Wt, int K, int N) {
  int idx = blockIdx.x * 256 + threadIdx.x;
  if (idx < K * N) {
    int k = idx / N, n = idx % N;
    Wt[(size_t)n * K + k] = (_Float16)W[idx];
  }
}

// ---------------------------------------------------------------------------
// RoPE (rotate_every_two, emb = concat(freqs,freqs)) + per-head repack.
//   Qh,Kh: [h][t][d] f16 (post-RoPE);  Vt: [h][d][t] f16 (transposed for P@V)
// ---------------------------------------------------------------------------
__global__ __launch_bounds__(256)
void rope_pack(const float* __restrict__ qkv, _Float16* __restrict__ Qh,
               _Float16* __restrict__ Kh, _Float16* __restrict__ Vt) {
  int idx = blockIdx.x * 256 + threadIdx.x;      // over T*C
  int t = idx >> 10;
  int c = idx & 1023;
  int h = c >> 6, d = c & 63;

  const float* row = qkv + (size_t)t * (3 * CC);
  float q  = row[c],          k  = row[CC + c],  v = row[2 * CC + c];
  int   cp = (c & ~63) | (d ^ 1);                // partner feature (d^1) in same head
  float qp = row[cp],         kp = row[CC + cp];

  // theta_d = t * 10000^(-(d mod 32)/32)
  float j     = (float)(d & 31);
  float invf  = __expf(j * (-9.210340371976184f / 32.0f)); // ln(10000)
  float theta = (float)t * invf;
  float cs = cosf(theta), sn = sinf(theta);

  float qo, ko;
  if (d & 1) { qo = q * cs + qp * sn;  ko = k * cs + kp * sn; }   // rot[2i+1] = x[2i]
  else       { qo = q * cs - qp * sn;  ko = k * cs - kp * sn; }   // rot[2i]   = -x[2i+1]

  size_t td = ((size_t)h * TT + t) * HD + d;
  Qh[td] = (_Float16)qo;
  Kh[td] = (_Float16)ko;
  Vt[((size_t)h * HD + d) * TT + t] = (_Float16)v;
}

// ---------------------------------------------------------------------------
// Flash attention (causal), block-cooperative.
// Block = 128 threads = 4 waves sharing one 64-row Q tile of one head.
// KV blocks of 32 staged into double-buffered LDS via async global->LDS b128,
// overlapped with compute (ASYNCcnt). Each wave owns a 16-row strip:
//   S = Q K^T (4 WMMAs from LDS K tile), online softmax, P@V (4 WMMAs from
//   LDS V^T tile). P passes through per-wave LDS to convert C->A layout.
// ---------------------------------------------------------------------------
__global__ __launch_bounds__(128)
void flash_attn(const _Float16* __restrict__ Qh, const _Float16* __restrict__ Kh,
                const _Float16* __restrict__ Vt, _Float16* __restrict__ Yh) {
  __shared__ __align__(16) _Float16 ktile[2][32 * 64];  // [buf][t_local][d]
  __shared__ __align__(16) _Float16 vtile[2][64 * 32];  // [buf][d][t_local]
  __shared__ __align__(16) _Float16 pst[4][16 * 32];

  const int tid  = threadIdx.x;
  const int lane = tid & 31;
  const int w    = tid >> 5;
  const int h    = blockIdx.y;
  const int q0b  = blockIdx.x * 64;     // block's Q rows
  const int q0   = q0b + w * 16;        // this wave's Q rows

  const _Float16* Q = Qh + ((size_t)h * TT + q0) * HD;
  const _Float16* K = Kh + (size_t)h * TT * HD;     // [t][d]
  const _Float16* V = Vt + (size_t)h * HD * TT;     // [d][t]
  _Float16* P = pst[w];

  v16h qa0 = load_a_frag(Q, HD, 0, 0,  lane);
  v16h qa1 = load_a_frag(Q, HD, 0, 32, lane);

  float mrow[8], lrow[8];
  v8f acc[4] = {};
#pragma unroll
  for (int r = 0; r < 8; ++r) { mrow[r] = -INFINITY; lrow[r] = 0.0f; }

  const float sc = 0.125f;              // 1/sqrt(64)
  const int nl = lane & 15;
  const int mh = (lane >> 4) << 3;

  const int nblk = q0b / 32 + 2;        // KV blocks this block must visit

  // stage KV block `it` into buffer `buf`: 4 async b128 per lane (2 K + 2 V)
  auto stage = [&](int it, int buf) {
    const int k0 = it * 32;
    unsigned kb = (unsigned)(size_t)&ktile[buf][0];
    unsigned vb = (unsigned)(size_t)&vtile[buf][0];
#pragma unroll
    for (int s = 0; s < 2; ++s) {
      int ck = tid + s * 128;           // K tile: 256 chunks of 16B, [t][d]
      async_b128(kb + ck * 16, K + (size_t)(k0 + (ck >> 3)) * HD + (ck & 7) * 8);
      int cv = tid + s * 128;           // V tile: 256 chunks of 16B, [d][t]
      async_b128(vb + cv * 16, V + (size_t)(cv >> 1) * TT + k0 + (cv & 1) * 8);
    }
  };

  stage(0, 0);
  for (int it = 0; it < nblk; ++it) {
    const int buf = it & 1;
    const int k0  = it * 32;
    if (it + 1 < nblk) {
      stage(it + 1, buf ^ 1);                          // prefetch next buffer
      asm volatile("s_wait_asynccnt 0x4" ::: "memory"); // current buffer done
    } else {
      asm volatile("s_wait_asynccnt 0x0" ::: "memory");
    }
    __syncthreads();

    if (k0 <= q0 + 15) {                // wave-uniform causal trip guard
      const _Float16* kt = ktile[buf];
      const _Float16* vt = vtile[buf];

      // ---- S = Q K^T, 16x32 tile as two 16x16 C-frags (K from LDS)
      v8f s0 = {}, s1 = {};
      s0 = wmma_f16(qa0, load_b_frag(kt, HD, 0,  0,  lane), s0);
      s0 = wmma_f16(qa1, load_b_frag(kt, HD, 0,  32, lane), s0);
      s1 = wmma_f16(qa0, load_b_frag(kt, HD, 16, 0,  lane), s1);
      s1 = wmma_f16(qa1, load_b_frag(kt, HD, 16, 32, lane), s1);

      // ---- scale + causal mask + row max
      const bool diag = (k0 + 31 > q0);
      float rmax[8];
#pragma unroll
      for (int r = 0; r < 8; ++r) {
        float v0 = s0[r] * sc, v1 = s1[r] * sc;
        if (diag) {
          int qt = q0 + mh + r;
          if (k0 + nl      > qt) v0 = -INFINITY;
          if (k0 + 16 + nl > qt) v1 = -INFINITY;
        }
        s0[r] = v0; s1[r] = v1;
        rmax[r] = fmaxf(v0, v1);
      }
#pragma unroll
      for (int off = 1; off < 16; off <<= 1)
#pragma unroll
        for (int r = 0; r < 8; ++r)
          rmax[r] = fmaxf(rmax[r], __shfl_xor(rmax[r], off, 32));

      // ---- online softmax update
      float corr[8], rsum[8];
#pragma unroll
      for (int r = 0; r < 8; ++r) {
        float mn = fmaxf(mrow[r], rmax[r]);
        corr[r] = __expf(mrow[r] - mn);
        mrow[r] = mn;
        float p0 = __expf(s0[r] - mn);
        float p1 = __expf(s1[r] - mn);
        s0[r] = p0; s1[r] = p1;
        rsum[r] = p0 + p1;
      }
#pragma unroll
      for (int off = 1; off < 16; off <<= 1)
#pragma unroll
        for (int r = 0; r < 8; ++r)
          rsum[r] += __shfl_xor(rsum[r], off, 32);
#pragma unroll
      for (int r = 0; r < 8; ++r) lrow[r] = lrow[r] * corr[r] + rsum[r];
#pragma unroll
      for (int j = 0; j < 4; ++j)
#pragma unroll
        for (int r = 0; r < 8; ++r) acc[j][r] *= corr[r];

      // ---- stage P (C-layout -> LDS row-major 16x32)
#pragma unroll
      for (int r = 0; r < 8; ++r) {
        P[(mh + r) * 32 + nl]      = (_Float16)s0[r];
        P[(mh + r) * 32 + 16 + nl] = (_Float16)s1[r];
      }
      asm volatile("s_wait_dscnt 0" ::: "memory");   // per-wave DS write->read order

      // ---- P @ V : A-frag from LDS, B-frags from LDS V^T tile
      v16h pa = load_a_frag(P, 32, 0, 0, lane);
#pragma unroll
      for (int j = 0; j < 4; ++j)
        acc[j] = wmma_f16(pa, load_b_frag(vt, 32, 16 * j, 0, lane), acc[j]);
    }
    __syncthreads();   // protect double buffer from next iteration's staging
  }

  // ---- normalize + store to y[t][h*64+d] (f16, feeds projection GEMM)
#pragma unroll
  for (int j = 0; j < 4; ++j)
#pragma unroll
    for (int r = 0; r < 8; ++r) {
      int row = q0 + mh + r;
      int col = h * HD + 16 * j + nl;
      Yh[(size_t)row * CC + col] = (_Float16)(acc[j][r] / lrow[r]);
    }
}

// ---------------------------------------------------------------------------
// Host launcher
// ---------------------------------------------------------------------------
extern "C" void kernel_launch(void* const* d_in, const int* in_sizes, int n_in,
                              void* d_out, int out_size, void* d_ws, size_t ws_size,
                              hipStream_t stream) {
  (void)in_sizes; (void)n_in; (void)out_size; (void)ws_size;
  const float* x     = (const float*)d_in[0];
  const float* Wqkv  = (const float*)d_in[1];
  const float* bqkv  = (const float*)d_in[2];
  const float* Wproj = (const float*)d_in[3];
  const float* bproj = (const float*)d_in[4];
  float* out = (float*)d_out;

  // workspace layout (96 MB total)
  char* ws = (char*)d_ws;
  float*    qkv    = (float*)ws;                               // 4096*3072*4 = 48 MB
  _Float16* Qh     = (_Float16*)(ws + (size_t)50331648);       // 8 MB
  _Float16* Kh     = Qh + (size_t)NH * TT * HD;                // 8 MB
  _Float16* Vt     = Kh + (size_t)NH * TT * HD;                // 8 MB
  _Float16* Yh     = Vt + (size_t)NH * TT * HD;                // 8 MB
  _Float16* xh     = Yh + (size_t)TT * CC;                     // 8 MB
  _Float16* Wqkvt  = xh + (size_t)TT * CC;                     // 6 MB
  _Float16* Wprojt = Wqkvt + (size_t)CC * 3 * CC;              // 2 MB

  // 1) precision conversion (+ weight transpose for contiguous B-fragments)
  convert_f16   <<<(TT * CC) / 256, 256, 0, stream>>>(x, xh, TT * CC);
  convert_tr_f16<<<(CC * 3 * CC) / 256, 256, 0, stream>>>(Wqkv, Wqkvt, CC, 3 * CC);
  convert_tr_f16<<<(CC * CC) / 256, 256, 0, stream>>>(Wproj, Wprojt, CC, CC);

  // 2) qkv = x @ Wqkv + bqkv    (M=4096, N=3072, K=1024)
  gemm_bias_f16<<<dim3(TT / 128, (3 * CC) / 128), 256, 0, stream>>>(
      xh, Wqkvt, bqkv, qkv, TT, 3 * CC, CC);

  // 3) RoPE + per-head repack (V transposed)
  rope_pack<<<(TT * CC) / 256, 256, 0, stream>>>(qkv, Qh, Kh, Vt);

  // 4) causal flash attention (async KV staging, double-buffered)
  flash_attn<<<dim3(TT / 64, NH), 128, 0, stream>>>(Qh, Kh, Vt, Yh);

  // 5) out = y @ Wproj + bproj  (M=4096, N=1024, K=1024)
  gemm_bias_f16<<<dim3(TT / 128, CC / 128), 256, 0, stream>>>(
      Yh, Wprojt, bproj, out, TT, CC, CC);
}
